// MiniMoE_47665547051338
// MI455X (gfx1250) — compile-verified, matrix operand order checked by reading, not run on hardware
//
#include <hip/hip_runtime.h>
#include <hip/hip_bf16.h>
#include <cstddef>

#define TOK   1024
#define DIM   512
#define NEXP  8
#define EDIM  1024
#define NMIC  325
#define HMIC  256

typedef __bf16 bf16_t;
typedef __attribute__((ext_vector_type(16))) __bf16 v16bf;
typedef __attribute__((ext_vector_type(8)))  float  v8f;
typedef __attribute__((ext_vector_type(4)))  int    v4i;

#define GLOBAL_AS __attribute__((address_space(1)))
#define LDS_AS    __attribute__((address_space(3)))

#if __has_builtin(__builtin_amdgcn_global_load_async_to_lds_b128) && \
    __has_builtin(__builtin_amdgcn_s_wait_asynccnt)
#define USE_ASYNC_LDS 1
#else
#define USE_ASYNC_LDS 0
#endif

__device__ __forceinline__ v8f wmma_bf16(v16bf a, v16bf b, v8f c) {
  return __builtin_amdgcn_wmma_f32_16x16x32_bf16(
      false, a, false, b, (short)0, c, false, false);
}

__device__ __forceinline__ float gelu_f(float x) {
  return 0.5f * x * (1.0f + erff(x * 0.70710678118654752440f));
}

// 16-byte global -> LDS copy, async (ASYNCcnt / global_load_async_to_lds_b128)
// if available.
__device__ __forceinline__ void copy16_to_lds(const bf16_t* g, bf16_t* l) {
#if USE_ASYNC_LDS
  void* gv = (void*)g;  // strip const in generic AS first
  __builtin_amdgcn_global_load_async_to_lds_b128(
      (GLOBAL_AS v4i*)gv, (LDS_AS v4i*)l, 0, 0);
#else
  *(uint4*)l = *(const uint4*)g;
#endif
}
__device__ __forceinline__ void async_wait_all() {
#if USE_ASYNC_LDS
  __builtin_amdgcn_s_wait_asynccnt(0);
#endif
}

// Fragment: lane l -> row (l&15); chunks at K = (l>>4)*8 and +16.
__device__ __forceinline__ v16bf frag_from(const bf16_t* p) {
  v16bf a;
#pragma unroll
  for (int i = 0; i < 8; ++i) a[i] = p[i];
#pragma unroll
  for (int i = 0; i < 8; ++i) a[8 + i] = p[16 + i];
  return a;
}

__device__ __forceinline__ v16bf load_frag_f32(const float* base, int stride,
                                               int row0, int k0, int lane) {
  int r = lane & 15, g = lane >> 4;
  const float* p = base + (size_t)(row0 + r) * stride + k0 + g * 8;
  v16bf a;
#pragma unroll
  for (int i = 0; i < 8; ++i) a[i] = (bf16_t)p[i];
#pragma unroll
  for (int i = 0; i < 8; ++i) a[8 + i] = (bf16_t)p[16 + i];
  return a;
}

// ---------------- fp32 -> bf16 bulk convert ----------------
__global__ void k_cvt(const float* __restrict__ src, bf16_t* __restrict__ dst,
                      int n) {
  int i = (blockIdx.x * blockDim.x + threadIdx.x) * 4;
  if (i >= n) return;
  float4 v = *(const float4*)(src + i);
  dst[i + 0] = (bf16_t)v.x;
  dst[i + 1] = (bf16_t)v.y;
  dst[i + 2] = (bf16_t)v.z;
  dst[i + 3] = (bf16_t)v.w;
}

// ---------------- zero scratch ----------------
__global__ void k_zero(float* mo, int* counts) {
  int i = blockIdx.x * blockDim.x + threadIdx.x;
  if (i < TOK * DIM) mo[i] = 0.0f;
  if (i < NMIC) counts[i] = 0;
}

// ---------------- expert router: top-2 of 8 ----------------
__global__ __launch_bounds__(256) void k_router(const float* __restrict__ x,
                                                const float* __restrict__ rw,
                                                const float* __restrict__ rb,
                                                float* __restrict__ comb) {
  int t = blockIdx.x * blockDim.x + threadIdx.x;
  if (t >= TOK) return;
  const float* xr = x + (size_t)t * DIM;
  float lg[NEXP];
#pragma unroll
  for (int e = 0; e < NEXP; ++e) {
    const float* w = rw + (size_t)e * DIM;
    float s = 0.f;
    for (int k = 0; k < DIM; ++k) s += xr[k] * w[k];
    lg[e] = s + rb[e];
  }
  float mx = lg[0];
#pragma unroll
  for (int e = 1; e < NEXP; ++e) mx = fmaxf(mx, lg[e]);
  float p[NEXP], den = 0.f;
#pragma unroll
  for (int e = 0; e < NEXP; ++e) { p[e] = expf(lg[e] - mx); den += p[e]; }
#pragma unroll
  for (int e = 0; e < NEXP; ++e) p[e] /= den;
  int i0 = 0;
#pragma unroll
  for (int e = 1; e < NEXP; ++e) if (p[e] > p[i0]) i0 = e;
  int i1 = (i0 == 0) ? 1 : 0;
#pragma unroll
  for (int e = 0; e < NEXP; ++e) if (e != i0 && p[e] > p[i1]) i1 = e;
  float s2 = p[i0] + p[i1] + 1e-8f;
#pragma unroll
  for (int e = 0; e < NEXP; ++e)
    comb[t * NEXP + e] = (e == i0) ? p[i0] / s2 : (e == i1) ? p[i1] / s2 : 0.f;
}

// ------- expert GEMM1: H = gelu(X*W1^T + b1); 32x32 tile/wave, async A tile -------
__global__ __launch_bounds__(256) void k_exp1(const bf16_t* __restrict__ xb,
                                              const bf16_t* __restrict__ w1b,
                                              const float* __restrict__ b1,
                                              bf16_t* __restrict__ Hbuf) {
  __shared__ bf16_t Alds[32 * 32];
  int lane = threadIdx.x & 31, wv = threadIdx.x >> 5;
  int r = lane & 15, g = lane >> 4;
  int mrow0 = blockIdx.x * 32;
  int e = blockIdx.z;
  int n0 = (blockIdx.y * 16 + wv * 2) * 16;  // first of two 16-col tiles
  const bf16_t* B = w1b + (size_t)e * EDIM * DIM;
  v8f c[2][2] = {};
  for (int k0 = 0; k0 < DIM; k0 += 32) {
    __syncthreads();
    if (threadIdx.x < 128) {
      int row = threadIdx.x >> 2, ch = threadIdx.x & 3;
      copy16_to_lds(xb + (size_t)(mrow0 + row) * DIM + k0 + ch * 8,
                    &Alds[row * 32 + ch * 8]);
    }
    async_wait_all();
    __syncthreads();
    v16bf a0 = frag_from(&Alds[r * 32 + g * 8]);
    v16bf a1 = frag_from(&Alds[(16 + r) * 32 + g * 8]);
    v16bf b0 = frag_from(B + (size_t)(n0 + r) * DIM + k0 + g * 8);
    v16bf b1 = frag_from(B + (size_t)(n0 + 16 + r) * DIM + k0 + g * 8);
    c[0][0] = wmma_bf16(a0, b0, c[0][0]);
    c[0][1] = wmma_bf16(a0, b1, c[0][1]);
    c[1][0] = wmma_bf16(a1, b0, c[1][0]);
    c[1][1] = wmma_bf16(a1, b1, c[1][1]);
  }
#pragma unroll
  for (int mi = 0; mi < 2; ++mi)
#pragma unroll
    for (int ni = 0; ni < 2; ++ni) {
      int col = n0 + ni * 16 + r;
      float bias = b1[(size_t)e * EDIM + col];
#pragma unroll
      for (int v = 0; v < 8; ++v) {
        int row = mrow0 + mi * 16 + v + 8 * g;
        Hbuf[((size_t)e * TOK + row) * EDIM + col] =
            (bf16_t)gelu_f(c[mi][ni][v] + bias);
      }
    }
}

// ---- expert GEMM2 fused combine: eo = sum_e comb[t,e]*(H[e]*W2^T + b2) ----
__global__ __launch_bounds__(256) void k_exp2(const bf16_t* __restrict__ Hbuf,
                                              const bf16_t* __restrict__ w2b,
                                              const float* __restrict__ b2,
                                              const float* __restrict__ comb,
                                              float* __restrict__ eo_f,
                                              bf16_t* __restrict__ eo_b) {
  __shared__ bf16_t Alds[32 * 32];
  int lane = threadIdx.x & 31, wv = threadIdx.x >> 5;
  int r = lane & 15, g = lane >> 4;
  int mrow0 = blockIdx.x * 32;
  int n0 = (blockIdx.y * 16 + wv * 2) * 16;
  float acc[2][2][8] = {};
  for (int e = 0; e < NEXP; ++e) {
    const bf16_t* A = Hbuf + (size_t)e * TOK * EDIM;
    const bf16_t* B = w2b + (size_t)e * DIM * EDIM;
    v8f c[2][2] = {};
    for (int k0 = 0; k0 < EDIM; k0 += 32) {
      __syncthreads();
      if (threadIdx.x < 128) {
        int row = threadIdx.x >> 2, ch = threadIdx.x & 3;
        copy16_to_lds(A + (size_t)(mrow0 + row) * EDIM + k0 + ch * 8,
                      &Alds[row * 32 + ch * 8]);
      }
      async_wait_all();
      __syncthreads();
      v16bf a0 = frag_from(&Alds[r * 32 + g * 8]);
      v16bf a1 = frag_from(&Alds[(16 + r) * 32 + g * 8]);
      v16bf b0 = frag_from(B + (size_t)(n0 + r) * EDIM + k0 + g * 8);
      v16bf b1 = frag_from(B + (size_t)(n0 + 16 + r) * EDIM + k0 + g * 8);
      c[0][0] = wmma_bf16(a0, b0, c[0][0]);
      c[0][1] = wmma_bf16(a0, b1, c[0][1]);
      c[1][0] = wmma_bf16(a1, b0, c[1][0]);
      c[1][1] = wmma_bf16(a1, b1, c[1][1]);
    }
#pragma unroll
    for (int ni = 0; ni < 2; ++ni) {
      int col = n0 + ni * 16 + r;
      float bias = b2[(size_t)e * DIM + col];
#pragma unroll
      for (int mi = 0; mi < 2; ++mi)
#pragma unroll
        for (int v = 0; v < 8; ++v) {
          int row = mrow0 + mi * 16 + v + 8 * g;
          acc[mi][ni][v] += comb[row * NEXP + e] * (c[mi][ni][v] + bias);
        }
    }
  }
#pragma unroll
  for (int mi = 0; mi < 2; ++mi)
#pragma unroll
    for (int ni = 0; ni < 2; ++ni) {
      int col = n0 + ni * 16 + r;
#pragma unroll
      for (int v = 0; v < 8; ++v) {
        int row = mrow0 + mi * 16 + v + 8 * g;
        eo_f[(size_t)row * DIM + col] = acc[mi][ni][v];
        eo_b[(size_t)row * DIM + col] = (bf16_t)acc[mi][ni][v];
      }
    }
}

// ---------------- micro router: top-8 of 325 (one wave per token) ----------------
__global__ __launch_bounds__(256) void k_mrouter(const float* __restrict__ eo,
                                                 const float* __restrict__ mw,
                                                 const float* __restrict__ mb,
                                                 int* __restrict__ mtopi,
                                                 float* __restrict__ mtopp,
                                                 int* __restrict__ counts) {
  int lane = threadIdx.x & 31;
  int t = blockIdx.x * 8 + (threadIdx.x >> 5);
  if (t >= TOK) return;
  const float* er = eo + (size_t)t * DIM;
  float p[11];
  int mArr[11];
#pragma unroll
  for (int i = 0; i < 11; ++i) {
    int m = lane + 32 * i;
    mArr[i] = m;
    if (m < NMIC) {
      const float* w = mw + (size_t)m * DIM;
      float s = 0.f;
      for (int k = 0; k < DIM; ++k) s += er[k] * w[k];
      p[i] = s + mb[m];
    } else p[i] = -1e30f;
  }
  float mx = -1e30f;
#pragma unroll
  for (int i = 0; i < 11; ++i) mx = fmaxf(mx, p[i]);
  for (int o = 16; o; o >>= 1) mx = fmaxf(mx, __shfl_xor(mx, o, 32));
  float sum = 0.f;
#pragma unroll
  for (int i = 0; i < 11; ++i) {
    p[i] = (mArr[i] < NMIC) ? expf(p[i] - mx) : 0.f;
    sum += p[i];
  }
  for (int o = 16; o; o >>= 1) sum += __shfl_xor(sum, o, 32);
  float inv = 1.f / sum;
#pragma unroll
  for (int i = 0; i < 11; ++i) p[i] *= inv;
  float tp[8]; int ti[8];
  for (int k = 0; k < 8; ++k) {
    float bv = -1.f; int bl = 0;
#pragma unroll
    for (int i = 0; i < 11; ++i) if (p[i] > bv) { bv = p[i]; bl = i; }
    float v = bv; int mi = mArr[bl];
    for (int o = 16; o; o >>= 1) {
      float ov = __shfl_xor(v, o, 32);
      int om = __shfl_xor(mi, o, 32);
      if (ov > v || (ov == v && om < mi)) { v = ov; mi = om; }
    }
    tp[k] = v; ti[k] = mi;
    if (mArr[bl] == mi) p[bl] = -1.f;
  }
  float s8 = 1e-8f;
#pragma unroll
  for (int k = 0; k < 8; ++k) s8 += tp[k];
  if (lane == 0) {
#pragma unroll
    for (int k = 0; k < 8; ++k) {
      mtopi[t * 8 + k] = ti[k];
      mtopp[t * 8 + k] = tp[k] / s8;
      atomicAdd(&counts[ti[k]], 1);
    }
  }
}

// ---------------- exclusive scan over 325 counts ----------------
__global__ __launch_bounds__(512) void k_scan(const int* __restrict__ counts,
                                              int* __restrict__ offsets,
                                              int* __restrict__ cursor) {
  __shared__ int s[512];
  int i = threadIdx.x;
  int c = (i < NMIC) ? counts[i] : 0;
  s[i] = c;
  __syncthreads();
  for (int d = 1; d < 512; d <<= 1) {
    int v = (i >= d) ? s[i - d] : 0;
    __syncthreads();
    s[i] += v;
    __syncthreads();
  }
  if (i < NMIC) { offsets[i] = s[i] - c; cursor[i] = 0; }
}

// ---------------- scatter (token, weight) pairs into per-micro bins ----------------
__global__ void k_scatter(const int* __restrict__ mtopi,
                          const float* __restrict__ mtopp,
                          const int* __restrict__ offsets,
                          int* __restrict__ cursor,
                          int* __restrict__ tok, float* __restrict__ wt) {
  int i = blockIdx.x * blockDim.x + threadIdx.x;
  if (i >= TOK * 8) return;
  int m = mtopi[i];
  int pos = atomicAdd(&cursor[m], 1);
  int j = offsets[m] + pos;
  tok[j] = i >> 3;
  wt[j] = mtopp[i];
}

// ----- per-micro block: WMMA FFN over token group + LN + atomic accumulate -----
__global__ __launch_bounds__(256) void k_micro(const float* __restrict__ eo_f,
                                               const bf16_t* __restrict__ eo_b,
                                               const float* __restrict__ mw1,
                                               const float* __restrict__ mb1,
                                               const float* __restrict__ mw2,
                                               const float* __restrict__ mb2,
                                               const float* __restrict__ mg,
                                               const float* __restrict__ mbeta,
                                               const int* __restrict__ counts,
                                               const int* __restrict__ offsets,
                                               const int* __restrict__ tok,
                                               const float* __restrict__ wt,
                                               float* __restrict__ mo) {
  __shared__ bf16_t hm[16 * HMIC];
  __shared__ float ym[16 * DIM];
  __shared__ int s_tok[16];
  __shared__ float s_w[16];
  int m = blockIdx.x;
  int nt = counts[m];
  if (nt == 0) return;
  int off = offsets[m];
  int lane = threadIdx.x & 31, wv = threadIdx.x >> 5;
  int r = lane & 15, g = lane >> 4;
  const float* W1 = mw1 + (size_t)m * HMIC * DIM;
  const float* W2 = mw2 + (size_t)m * DIM * HMIC;

  for (int mt0 = 0; mt0 < nt; mt0 += 16) {
    if (threadIdx.x < 16) {
      int idx = mt0 + threadIdx.x;
      if (idx < nt) { s_tok[threadIdx.x] = tok[off + idx]; s_w[threadIdx.x] = wt[off + idx]; }
      else          { s_tok[threadIdx.x] = tok[off];       s_w[threadIdx.x] = 0.f; }
    }
    __syncthreads();

    for (int sub = 0; sub < 2; ++sub) {
      int ntile = wv + sub * 8;
      v8f c = {};
      for (int k0 = 0; k0 < DIM; k0 += 32) {
        v16bf a = frag_from(eo_b + (size_t)s_tok[r] * DIM + k0 + g * 8);
        v16bf b = load_frag_f32(W1, DIM, ntile * 16, k0, lane);
        c = wmma_bf16(a, b, c);
      }
      int col = ntile * 16 + r;
      float bias = mb1[(size_t)m * HMIC + col];
#pragma unroll
      for (int v = 0; v < 8; ++v)
        hm[(v + 8 * g) * HMIC + col] = (bf16_t)gelu_f(c[v] + bias);
    }
    __syncthreads();

    for (int sub = 0; sub < 4; ++sub) {
      int ntile = wv * 4 + sub;
      v8f c = {};
      for (int k0 = 0; k0 < HMIC; k0 += 32) {
        v16bf a = frag_from(hm + r * HMIC + k0 + g * 8);
        v16bf b = load_frag_f32(W2, HMIC, ntile * 16, k0, lane);
        c = wmma_bf16(a, b, c);
      }
      int col = ntile * 16 + r;
      float bias = mb2[(size_t)m * DIM + col];
#pragma unroll
      for (int v = 0; v < 8; ++v)
        ym[(v + 8 * g) * DIM + col] = c[v] + bias;
    }
    __syncthreads();

    for (int rr = wv * 2; rr < wv * 2 + 2; ++rr) {
      float pw = s_w[rr];
      int tk = s_tok[rr];
      float vals[16], sum = 0.f, sq = 0.f;
#pragma unroll
      for (int j = 0; j < 16; ++j) {
        int cidx = lane + 32 * j;
        float v = eo_f[(size_t)tk * DIM + cidx] + ym[rr * DIM + cidx];
        vals[j] = v; sum += v; sq += v * v;
      }
      for (int o = 16; o; o >>= 1) {
        sum += __shfl_xor(sum, o, 32);
        sq += __shfl_xor(sq, o, 32);
      }
      float mu = sum * (1.f / DIM);
      float var = sq * (1.f / DIM) - mu * mu;
      float rstd = rsqrtf(var + 1e-5f);
      if (pw != 0.f) {
#pragma unroll
        for (int j = 0; j < 16; ++j) {
          int cidx = lane + 32 * j;
          float n = (vals[j] - mu) * rstd * mg[(size_t)m * DIM + cidx] +
                    mbeta[(size_t)m * DIM + cidx];
          atomicAdd(&mo[(size_t)tk * DIM + cidx], pw * n);
        }
      }
    }
    __syncthreads();
  }
}

// ---------------- final LN: out = LN(eo + 0.1*mo) ----------------
__global__ __launch_bounds__(256) void k_final(const float* __restrict__ eo,
                                               const float* __restrict__ mo,
                                               const float* __restrict__ gg,
                                               const float* __restrict__ bb,
                                               float* __restrict__ out) {
  int lane = threadIdx.x & 31;
  int t = blockIdx.x * 8 + (threadIdx.x >> 5);
  if (t >= TOK) return;
  float vals[16], sum = 0.f, sq = 0.f;
#pragma unroll
  for (int j = 0; j < 16; ++j) {
    int c = lane + 32 * j;
    float v = eo[(size_t)t * DIM + c] + 0.1f * mo[(size_t)t * DIM + c];
    vals[j] = v; sum += v; sq += v * v;
  }
  for (int o = 16; o; o >>= 1) {
    sum += __shfl_xor(sum, o, 32);
    sq += __shfl_xor(sq, o, 32);
  }
  float mu = sum * (1.f / DIM);
  float var = sq * (1.f / DIM) - mu * mu;
  float rstd = rsqrtf(var + 1e-5f);
#pragma unroll
  for (int j = 0; j < 16; ++j) {
    int c = lane + 32 * j;
    out[(size_t)t * DIM + c] = (vals[j] - mu) * rstd * gg[c] + bb[c];
  }
}

extern "C" void kernel_launch(void* const* d_in, const int* in_sizes, int n_in,
                              void* d_out, int out_size, void* d_ws, size_t ws_size,
                              hipStream_t stream) {
  const float* x        = (const float*)d_in[0];
  const float* router_w = (const float*)d_in[1];
  const float* router_b = (const float*)d_in[2];
  const float* exp_w1   = (const float*)d_in[3];
  const float* exp_b1   = (const float*)d_in[4];
  const float* exp_w2   = (const float*)d_in[5];
  const float* exp_b2   = (const float*)d_in[6];
  const float* mrouter_w= (const float*)d_in[7];
  const float* mrouter_b= (const float*)d_in[8];
  const float* mic_w1   = (const float*)d_in[9];
  const float* mic_b1   = (const float*)d_in[10];
  const float* mic_w2   = (const float*)d_in[11];
  const float* mic_b2   = (const float*)d_in[12];
  const float* mic_g    = (const float*)d_in[13];
  const float* mic_beta = (const float*)d_in[14];
  const float* norm_g   = (const float*)d_in[15];
  const float* norm_b   = (const float*)d_in[16];
  float* out = (float*)d_out;

  char* w = (char*)d_ws;
  size_t o = 0;
  auto carve = [&](size_t bytes) -> char* {
    char* r = w + o;
    o += (bytes + 255) & ~(size_t)255;
    return r;
  };
  float*  comb   = (float*)carve((size_t)TOK * NEXP * 4);
  bf16_t* Hbuf   = (bf16_t*)carve((size_t)NEXP * TOK * EDIM * 2);
  float*  eo_f   = (float*)carve((size_t)TOK * DIM * 4);
  bf16_t* eo_b   = (bf16_t*)carve((size_t)TOK * DIM * 2);
  int*    mtopi  = (int*)carve((size_t)TOK * 8 * 4);
  float*  mtopp  = (float*)carve((size_t)TOK * 8 * 4);
  int*    counts = (int*)carve((size_t)NMIC * 4);
  int*    offs   = (int*)carve((size_t)NMIC * 4);
  int*    cursor = (int*)carve((size_t)NMIC * 4);
  int*    tokl   = (int*)carve((size_t)TOK * 8 * 4);
  float*  wtl    = (float*)carve((size_t)TOK * 8 * 4);
  float*  mo     = (float*)carve((size_t)TOK * DIM * 4);
  bf16_t* xb     = (bf16_t*)carve((size_t)TOK * DIM * 2);
  bf16_t* w1b    = (bf16_t*)carve((size_t)NEXP * EDIM * DIM * 2);
  bf16_t* w2b    = (bf16_t*)carve((size_t)NEXP * DIM * EDIM * 2);

  const int nX = TOK * DIM, nW = NEXP * EDIM * DIM;
  k_cvt<<<(nX / 4 + 255) / 256, 256, 0, stream>>>(x, xb, nX);
  k_cvt<<<(nW / 4 + 255) / 256, 256, 0, stream>>>(exp_w1, w1b, nW);
  k_cvt<<<(nW / 4 + 255) / 256, 256, 0, stream>>>(exp_w2, w2b, nW);

  k_zero<<<(TOK * DIM + 255) / 256, 256, 0, stream>>>(mo, counts);
  k_router<<<TOK / 256, 256, 0, stream>>>(x, router_w, router_b, comb);
  k_exp1<<<dim3(TOK / 32, EDIM / 256, NEXP), 256, 0, stream>>>(xb, w1b, exp_b1, Hbuf);
  k_exp2<<<dim3(TOK / 32, DIM / 256, 1), 256, 0, stream>>>(Hbuf, w2b, exp_b2, comb,
                                                           eo_f, eo_b);
  k_mrouter<<<TOK / 8, 256, 0, stream>>>(eo_f, mrouter_w, mrouter_b, mtopi, mtopp,
                                         counts);
  k_scan<<<1, 512, 0, stream>>>(counts, offs, cursor);
  k_scatter<<<(TOK * 8 + 255) / 256, 256, 0, stream>>>(mtopi, mtopp, offs, cursor,
                                                       tokl, wtl);
  k_micro<<<NMIC, 256, 0, stream>>>(eo_f, eo_b, mic_w1, mic_b1, mic_w2, mic_b2,
                                    mic_g, mic_beta, counts, offs, tokl, wtl, mo);
  k_final<<<TOK / 8, 256, 0, stream>>>(eo_f, mo, norm_g, norm_b, out);
}
